// MultiheadSelfAttention_37993280700994
// MI455X (gfx1250) — matrix operand order repaired and checked by reading, compiled, not run on hardware
//
#include <hip/hip_runtime.h>
#include <hip/hip_bf16.h>

#define B_ 4
#define S_ 2048
#define D_ 1024
#define H_ 16
#define DK_ 64

typedef __attribute__((ext_vector_type(16))) __bf16 v16bf;
typedef __attribute__((ext_vector_type(8)))  float  v8f;

union FragBF {
  v16bf v;
  unsigned int u[8];
  unsigned short h[16];
};

__device__ __forceinline__ unsigned short f2bf(float f) {
  union { float f; unsigned int u; } x; x.f = f;
  unsigned int u = x.u;
  u += 0x7FFFu + ((u >> 16) & 1u);   // round-to-nearest-even
  return (unsigned short)(u >> 16);
}
__device__ __forceinline__ float bf2f(unsigned short h) {
  union { unsigned int u; float f; } x; x.u = ((unsigned)h) << 16; return x.f;
}

// A-matrix (16-bit, 16x32) K index for VGPR v, lane-half h (ISA 7.12.2)
__device__ __forceinline__ int a_kbase(int v, int half) {
  return (v < 4) ? (half * 8 + 2 * v) : (16 + half * 8 + 2 * (v - 4));
}

// -------- CDNA5 async global->LDS (ASYNCcnt) ------------------------------
__device__ __forceinline__ unsigned lds_off(const void* p) {
  // AS(3) -> generic keeps the wave-relative LDS byte offset in the low 32 bits
  return (unsigned)(unsigned long long)p;
}
__device__ __forceinline__ void async_b128(unsigned ldsoff, const void* g) {
  asm volatile("global_load_async_to_lds_b128 %0, %1, off"
               :: "v"(ldsoff), "v"(g) : "memory");
}
__device__ __forceinline__ void wait_async_le8() {
  asm volatile("s_wait_asynccnt 0x8" ::: "memory");
}
__device__ __forceinline__ void wait_async_le4() {
  asm volatile("s_wait_asynccnt 0x4" ::: "memory");
}
__device__ __forceinline__ void wait_async_le0() {
  asm volatile("s_wait_asynccnt 0x0" ::: "memory");
}

// ---------------------------------------------------------------- convert
__global__ void cvt_f32_bf16(const float* __restrict__ src,
                             unsigned short* __restrict__ dst, int n) {
  int i = blockIdx.x * blockDim.x + threadIdx.x;
  if (i < n) dst[i] = f2bf(src[i]);
}

// ---------------------------------------------------------------- GEMM
// C[M][N] = A[M][K](bf16 rm) * Bt[N][K](bf16 rm)^T
// 256 thr (8 waves), 128x128 tile, wave = 32 rows x 64 cols (8 WMMA/iter),
// K-step 32, double-buffered LDS filled by async b128 loads.
#define GS 40   // LDS row stride in shorts (80B: 16B-aligned rows, conflict-free)

__device__ __forceinline__ void gemm_issue(
    const unsigned short* __restrict__ A, const unsigned short* __restrict__ Bt,
    unsigned short* As, unsigned short* Bs, int m0, int n0, int K, int k0, int t) {
#pragma unroll
  for (int p = 0; p < 2; ++p) {
    int li = t + p * 256;          // 0..511: 128 rows x 4 uint4
    int row = li >> 2, c4 = li & 3;
    async_b128(lds_off(&As[row * GS + c4 * 8]), A  + (size_t)(m0 + row) * K + k0 + c4 * 8);
    async_b128(lds_off(&Bs[row * GS + c4 * 8]), Bt + (size_t)(n0 + row) * K + k0 + c4 * 8);
  }
}

template <bool OUT_BF16>
__global__ __launch_bounds__(256) void gemm_bf16_wmma(
    const unsigned short* __restrict__ A, const unsigned short* __restrict__ Bt,
    void* __restrict__ Cout, int M, int N, int K) {
  __shared__ unsigned short As[2][128 * GS];
  __shared__ unsigned short Bs[2][128 * GS];
  const int t    = threadIdx.x;
  const int lane = t & 31;
  const int wave = t >> 5;           // 0..7
  const int half = lane >> 4;
  const int l16  = lane & 15;
  const int rw   = (wave & 3) * 32;  // wave row offset in tile (4 x 32 = 128)
  const int cw   = (wave >> 2) * 64; // wave col offset in tile (2 x 64 = 128)
  const int m0   = blockIdx.x * 128;
  const int n0   = blockIdx.y * 128;

  v8f acc[2][4];
#pragma unroll
  for (int rg = 0; rg < 2; ++rg)
#pragma unroll
    for (int ns = 0; ns < 4; ++ns)
#pragma unroll
      for (int r = 0; r < 8; ++r) acc[rg][ns][r] = 0.0f;

  gemm_issue(A, Bt, As[0], Bs[0], m0, n0, K, 0, t);

  int buf = 0;
  for (int k0 = 0; k0 < K; k0 += 32, buf ^= 1) {
    const bool more = (k0 + 32 < K);
    if (more) gemm_issue(A, Bt, As[buf ^ 1], Bs[buf ^ 1], m0, n0, K, k0 + 32, t);
    if (more) wait_async_le4(); else wait_async_le0();
    __syncthreads();

    FragBF af[2];
#pragma unroll
    for (int rg = 0; rg < 2; ++rg)
#pragma unroll
      for (int v = 0; v < 8; ++v)
        af[rg].u[v] = *(const unsigned int*)
            &As[buf][(rw + rg * 16 + l16) * GS + half * 16 + 2 * v];

#pragma unroll
    for (int ns = 0; ns < 4; ++ns) {
      FragBF bf;
#pragma unroll
      for (int v = 0; v < 8; ++v)
        bf.u[v] = *(const unsigned int*)
            &Bs[buf][(cw + ns * 16 + l16) * GS + half * 16 + 2 * v];
#pragma unroll
      for (int rg = 0; rg < 2; ++rg)
        acc[rg][ns] = __builtin_amdgcn_wmma_f32_16x16x32_bf16(
            false, af[rg].v, false, bf.v, (short)0, acc[rg][ns], false, false);
    }
    __syncthreads();   // read-done before next-iter overwrite of buf
  }

#pragma unroll
  for (int rg = 0; rg < 2; ++rg)
#pragma unroll
    for (int ns = 0; ns < 4; ++ns)
#pragma unroll
      for (int r = 0; r < 8; ++r) {
        int row = m0 + rw + rg * 16 + r + 8 * half;   // C layout: M = r + 8*half
        int col = n0 + cw + ns * 16 + l16;            //           N = lane%16
        if (OUT_BF16) ((unsigned short*)Cout)[(size_t)row * N + col] = f2bf(acc[rg][ns][r]);
        else          ((float*)Cout)[(size_t)row * N + col] = acc[rg][ns][r];
      }
}

// ---------------------------------------------------------------- RoPE + split
// qkvb[b][s][3*1024] bf16 -> roped Q,K bf16 [B,H,S,DK]; V^T bf16 [B,H,DK,S]
__global__ __launch_bounds__(256) void rope_split(
    const unsigned short* __restrict__ qkvb, const int* __restrict__ pos,
    unsigned short* __restrict__ Q, unsigned short* __restrict__ Kk,
    unsigned short* __restrict__ Vt) {
  __shared__ unsigned short Vl[64][72];   // [dk][s_local], 144B rows (16B-aligned)
  const int blk = blockIdx.x;
  const int sb  = blk & (S_ / 64 - 1);
  const int h   = (blk >> 5) & (H_ - 1);
  const int b   = blk >> 9;
  const int t   = threadIdx.x;
  const int i2  = t & 31;
  const int sl0 = t >> 5;    // 0..7

  const float freq = __expf(-(float)(2 * i2) * (1.0f / 64.0f) * 9.210340371976184f);

#pragma unroll
  for (int rep = 0; rep < 8; ++rep) {
    int sl = sl0 * 8 + rep;           // 0..63
    int s  = sb * 64 + sl;
    float p = (float)pos[s];
    float sn, c;
    __sincosf(p * freq, &sn, &c);
    size_t base = (size_t)(b * S_ + s) * 3072 + h * 64 + 2 * i2;
    float q1 = bf2f(qkvb[base]),        q2 = bf2f(qkvb[base + 1]);
    float k1 = bf2f(qkvb[base + 1024]), k2 = bf2f(qkvb[base + 1025]);
    size_t ho = ((size_t)(b * H_ + h) * S_ + s) * DK_ + 2 * i2;
    Q[ho]      = f2bf(q1 * c - q2 * sn);
    Q[ho + 1]  = f2bf(q1 * sn + q2 * c);
    Kk[ho]     = f2bf(k1 * c - k2 * sn);
    Kk[ho + 1] = f2bf(k1 * sn + k2 * c);
    Vl[2 * i2][sl]     = qkvb[base + 2048];
    Vl[2 * i2 + 1][sl] = qkvb[base + 2049];
  }
  __syncthreads();
#pragma unroll
  for (int p2 = 0; p2 < 2; ++p2) {
    int li = t + p2 * 256;            // 0..511: 64 dk rows x 8 uint4
    int dk = li >> 3, c8 = li & 7;
    uint4 dv = *(const uint4*)&Vl[dk][c8 * 8];
    *(uint4*)&Vt[((size_t)(b * H_ + h) * DK_ + dk) * S_ + sb * 64 + c8 * 8] = dv;
  }
}

// ---------------------------------------------------------------- flash attention
// 4 waves / block, 64 q rows; 64-key chunks double-buffered in LDS via async
// b128; exp2-domain online softmax; QK^T and P*V via WMMA (16/wave/chunk).
#define KS 72   // row stride (shorts): 144B rows, 16B-aligned, conflict-free

__device__ __forceinline__ void attn_issue(
    const unsigned short* __restrict__ Kk, const unsigned short* __restrict__ Vt,
    unsigned short* KsBuf, unsigned short* VsBuf, size_t bh, int kt, int t) {
#pragma unroll
  for (int p = 0; p < 4; ++p) {
    int li = t + p * 128;                 // 0..511: 64 rows x 8 uint4
    int row = li >> 3, c8 = li & 7;
    async_b128(lds_off(&KsBuf[row * KS + c8 * 8]),
               Kk + (bh * S_ + kt * 64 + row) * DK_ + c8 * 8);   // [key][dk]
    async_b128(lds_off(&VsBuf[row * KS + c8 * 8]),
               Vt + (bh * DK_ + row) * S_ + kt * 64 + c8 * 8);   // [dk][key]
  }
}

__global__ __launch_bounds__(128) void attn_fa(
    const unsigned short* __restrict__ Q, const unsigned short* __restrict__ Kk,
    const unsigned short* __restrict__ Vt, unsigned short* __restrict__ O) {
  __shared__ unsigned short Ks[2][64 * KS];
  __shared__ unsigned short Vs[2][64 * KS];
  __shared__ unsigned short Ps[4][16 * KS];
  const int t    = threadIdx.x;
  const int lane = t & 31;
  const int wave = t >> 5;
  const int half = lane >> 4;
  const int l16  = lane & 15;
  const int qb = blockIdx.x & (S_ / 64 - 1);
  const int h  = (blockIdx.x >> 5) & (H_ - 1);
  const int b  = blockIdx.x >> 9;
  const size_t bh = (size_t)(b * H_ + h);
  const int qrow0 = qb * 64 + wave * 16;      // this wave's first q row

  // Q fragments: A 16x64 = two 16x32 frags (one-time global gather)
  FragBF qf[2];
#pragma unroll
  for (int f = 0; f < 2; ++f)
#pragma unroll
    for (int v = 0; v < 8; ++v) {
      int kk = half * 16 + 2 * v + 32 * f;
      qf[f].u[v] = *(const unsigned int*)&Q[(bh * S_ + qrow0 + l16) * DK_ + kk];
    }

  v8f oacc[4];
  float mrow[8], lrow[8];
#pragma unroll
  for (int r = 0; r < 8; ++r) { mrow[r] = -3.0e38f; lrow[r] = 0.0f; }
#pragma unroll
  for (int ns = 0; ns < 4; ++ns)
#pragma unroll
    for (int r = 0; r < 8; ++r) oacc[ns][r] = 0.0f;

  const float SC2 = 0.125f * 1.4426950408889634f;   // (1/sqrt(dk)) * log2(e)
  const int nchunk = qb + 1;                        // 64-key chunks, uniform
  attn_issue(Kk, Vt, Ks[0], Vs[0], bh, 0, t);

  for (int kt = 0; kt < nchunk; ++kt) {
    const int cur = kt & 1;
    const bool more = (kt + 1 < nchunk);
    if (more) attn_issue(Kk, Vt, Ks[cur ^ 1], Vs[cur ^ 1], bh, kt + 1, t);
    if (more) wait_async_le8(); else wait_async_le0();
    __syncthreads();   // chunk kt resident in LDS for all waves

    // ---- S = Q K^T : four 16-key subtiles, K=64 via 2 chained WMMAs each
    v8f sc[4];
#pragma unroll
    for (int j = 0; j < 4; ++j) {
      v8f cz;
#pragma unroll
      for (int r = 0; r < 8; ++r) cz[r] = 0.0f;
#pragma unroll
      for (int f = 0; f < 2; ++f) {
        FragBF kfrag;
#pragma unroll
        for (int v = 0; v < 8; ++v)
          kfrag.u[v] = *(const unsigned int*)
              &Ks[cur][(j * 16 + l16) * KS + half * 16 + 2 * v + 32 * f];
        cz = __builtin_amdgcn_wmma_f32_16x16x32_bf16(
            false, qf[f].v, false, kfrag.v, (short)0, cz, false, false);
      }
      sc[j] = cz;
    }

    // ---- online softmax in exp2 domain; mask only on the diagonal chunk
    const bool diag = (kt == qb);
#pragma unroll
    for (int r = 0; r < 8; ++r) {
      int qrow = qrow0 + r + 8 * half;
#pragma unroll
      for (int j = 0; j < 4; ++j) {
        float val = sc[j][r] * SC2;
        if (diag) {
          int key = kt * 64 + j * 16 + l16;
          val = (key <= qrow) ? val : -3.0e38f;
        }
        sc[j][r] = val;
      }
      float mx = fmaxf(fmaxf(sc[0][r], sc[1][r]), fmaxf(sc[2][r], sc[3][r]));
      mx = fmaxf(mx, __shfl_xor(mx, 1));
      mx = fmaxf(mx, __shfl_xor(mx, 2));
      mx = fmaxf(mx, __shfl_xor(mx, 4));
      mx = fmaxf(mx, __shfl_xor(mx, 8));           // reduce over N within half
      float mnew  = fmaxf(mrow[r], mx);
      float alpha = exp2f(mrow[r] - mnew);
      float ls = 0.0f;
#pragma unroll
      for (int j = 0; j < 4; ++j) {
        float pj = exp2f(sc[j][r] - mnew);
        sc[j][r] = pj;
        ls += pj;
      }
      ls += __shfl_xor(ls, 1);
      ls += __shfl_xor(ls, 2);
      ls += __shfl_xor(ls, 4);
      ls += __shfl_xor(ls, 8);
      lrow[r] = lrow[r] * alpha + ls;
      mrow[r] = mnew;
#pragma unroll
      for (int ns = 0; ns < 4; ++ns) oacc[ns][r] *= alpha;
    }

    // ---- reshape P (C-layout, 16x64) -> two A-fragments via per-wave LDS
#pragma unroll
    for (int j = 0; j < 4; ++j)
#pragma unroll
      for (int r = 0; r < 8; ++r)
        Ps[wave][(r + 8 * half) * KS + j * 16 + l16] = f2bf(sc[j][r]);
    __syncthreads();   // conservative cross-lane LDS write->read ordering

    FragBF pf[2];
#pragma unroll
    for (int f = 0; f < 2; ++f)
#pragma unroll
      for (int v = 0; v < 8; ++v)
        pf[f].u[v] = *(const unsigned int*)
            &Ps[wave][l16 * KS + half * 16 + 2 * v + 32 * f];

    // ---- O += P * V   (K=64: two chained WMMAs per 16-col output subtile)
#pragma unroll
    for (int ns = 0; ns < 4; ++ns) {
#pragma unroll
      for (int f = 0; f < 2; ++f) {
        FragBF vf;
#pragma unroll
        for (int v = 0; v < 8; ++v)
          vf.u[v] = *(const unsigned int*)
              &Vs[cur][(ns * 16 + l16) * KS + half * 16 + 2 * v + 32 * f];
        oacc[ns] = __builtin_amdgcn_wmma_f32_16x16x32_bf16(
            false, pf[f].v, false, vf.v, (short)0, oacc[ns], false, false);
      }
    }
    __syncthreads();   // read-done before next-iter overwrite of cur buffers
  }

  // ---- epilogue: normalize, write bf16 [B,S,D]
#pragma unroll
  for (int ns = 0; ns < 4; ++ns)
#pragma unroll
    for (int r = 0; r < 8; ++r) {
      float val = oacc[ns][r] / lrow[r];
      int srow = qrow0 + r + 8 * half;
      O[(size_t)(b * S_ + srow) * D_ + h * 64 + ns * 16 + l16] = f2bf(val);
    }
}

// ---------------------------------------------------------------- launch
extern "C" void kernel_launch(void* const* d_in, const int* in_sizes, int n_in,
                              void* d_out, int out_size, void* d_ws, size_t ws_size,
                              hipStream_t stream) {
  (void)in_sizes; (void)n_in; (void)out_size; (void)ws_size;
  const float* x    = (const float*)d_in[0];
  const int*   pos  = (const int*)d_in[1];
  const float* Wqkv = (const float*)d_in[2];
  const float* Wo   = (const float*)d_in[3];
  float* out = (float*)d_out;

  char* w = (char*)d_ws;
  size_t off = 0;
  auto alloc = [&](size_t bytes) -> void* {
    void* p = w + off;
    off += (bytes + 255) & ~(size_t)255;
    return p;
  };
  const size_t MS = (size_t)B_ * S_;                 // 8192 rows
  unsigned short* xb    = (unsigned short*)alloc(MS * D_ * 2);
  unsigned short* wqkvb = (unsigned short*)alloc((size_t)3 * D_ * D_ * 2);
  unsigned short* wob   = (unsigned short*)alloc((size_t)D_ * D_ * 2);
  unsigned short* qkvb  = (unsigned short*)alloc(MS * 3 * D_ * 2);
  unsigned short* Qb    = (unsigned short*)alloc(MS * D_ * 2);
  unsigned short* Kb    = (unsigned short*)alloc(MS * D_ * 2);
  unsigned short* Vt    = (unsigned short*)alloc(MS * D_ * 2);
  unsigned short* Ob    = (unsigned short*)alloc(MS * D_ * 2);

  int nx = (int)(MS * D_);
  cvt_f32_bf16<<<(nx + 255) / 256, 256, 0, stream>>>(x, xb, nx);
  cvt_f32_bf16<<<(3 * D_ * D_ + 255) / 256, 256, 0, stream>>>(Wqkv, wqkvb, 3 * D_ * D_);
  cvt_f32_bf16<<<(D_ * D_ + 255) / 256, 256, 0, stream>>>(Wo, wob, D_ * D_);

  gemm_bf16_wmma<true><<<dim3((int)MS / 128, 3 * D_ / 128), 256, 0, stream>>>(
      xb, wqkvb, qkvb, (int)MS, 3 * D_, D_);

  rope_split<<<B_ * H_ * (S_ / 64), 256, 0, stream>>>(qkvb, pos, Qb, Kb, Vt);

  attn_fa<<<B_ * H_ * (S_ / 64), 128, 0, stream>>>(Qb, Kb, Vt, Ob);

  gemm_bf16_wmma<false><<<dim3((int)MS / 128, D_ / 128), 256, 0, stream>>>(
      Ob, wob, out, (int)MS, D_, D_);
}
